// MultiHeadAttention_25864293056986
// MI455X (gfx1250) — compile-verified, matrix operand order checked by reading, not run on hardware
//
#include <hip/hip_runtime.h>
#include <hip/hip_bf16.h>

#define D_  768
#define H_  12
#define DH_ 64
#define B_  16
#define N_  1024
#define BH_ (B_ * H_)   // 192

typedef __attribute__((ext_vector_type(16))) __bf16 v16bf;
typedef __attribute__((ext_vector_type(8)))  float  v8f;
typedef __attribute__((ext_vector_type(4)))  __bf16 v4bf;

union Frag {
    v16bf v;
    uint4 u[2];
};

union Pack8 {
    uint4  u;
    __bf16 h[8];
};

__device__ __forceinline__ v8f wmma_bf16(const Frag& a, const Frag& b, v8f c) {
    return __builtin_amdgcn_wmma_f32_16x16x32_bf16(
        /*neg_a=*/false, a.v, /*neg_b=*/false, b.v,
        /*c_mod=*/(short)0, c, /*reuse_a=*/false, /*reuse_b=*/false);
}

// Async global -> LDS copy of 16 bytes (CDNA5 GLOBAL_LOAD_ASYNC_TO_LDS_B128,
// tracked with ASYNCcnt).
__device__ __forceinline__ void async_copy16(void* lds, const void* gaddr) {
    unsigned int l = (unsigned int)(unsigned long long)lds;
    asm volatile("global_load_async_to_lds_b128 %0, %1, off"
                 :: "v"(l), "v"(gaddr) : "memory");
}

__device__ __forceinline__ void wait_async0() {
    asm volatile("s_wait_asynccnt 0x0" ::: "memory");
}

// ---------------------------------------------------------------------------
// Prep 1: x f32 -> bf16, 4 elements per thread
// ---------------------------------------------------------------------------
__global__ void __launch_bounds__(256) prep_x_kernel(const float* __restrict__ x,
                                                     __bf16* __restrict__ xb, int n4) {
    int i = blockIdx.x * blockDim.x + threadIdx.x;
    if (i >= n4) return;
    float4 f = ((const float4*)x)[i];
    v4bf o;
    o[0] = (__bf16)f.x; o[1] = (__bf16)f.y; o[2] = (__bf16)f.z; o[3] = (__bf16)f.w;
    ((v4bf*)xb)[i] = o;
}

// ---------------------------------------------------------------------------
// Prep 2: W[h][d][e] f32 -> Wt[m][h][e][d] bf16 (transposed: B-frag rows
// contiguous over K)
// ---------------------------------------------------------------------------
__global__ void __launch_bounds__(256) prep_w_kernel(const float* __restrict__ Wq,
                                                     const float* __restrict__ Wk,
                                                     const float* __restrict__ Wv,
                                                     __bf16* __restrict__ Wt, int total) {
    int i = blockIdx.x * blockDim.x + threadIdx.x;
    if (i >= total) return;
    int d = i % D_;
    int e = (i / D_) % DH_;
    int h = (i / (D_ * DH_)) % H_;
    int m = i / (D_ * DH_ * H_);
    const float* W = (m == 0) ? Wq : (m == 1) ? Wk : Wv;
    Wt[i] = (__bf16)W[((size_t)h * D_ + d) * DH_ + e];
}

// ---------------------------------------------------------------------------
// QKV projection: one wave per (token-block, matrix, head); 4 feature tiles
// per wave share one A fragment per K-chunk (4x A reuse).
//   q,k stored [bh][n][64] row-major bf16; v stored transposed [bh][64][n]
// ---------------------------------------------------------------------------
__global__ void __launch_bounds__(256) proj_qkv_kernel(
    const __bf16* __restrict__ xb, const __bf16* __restrict__ Wt,
    const float* __restrict__ bq, const float* __restrict__ bk,
    const float* __restrict__ bv,
    __bf16* __restrict__ qbuf, __bf16* __restrict__ kbuf, __bf16* __restrict__ vbuf)
{
    const int lane = threadIdx.x & 31;
    const int wv   = threadIdx.x >> 5;
    const int job  = blockIdx.x * 8 + wv;   // 0 .. 36863

    const int h  = job % H_;
    const int m  = (job / H_) % 3;
    const int tb = job / (H_ * 3);          // 16-token block id, 0..1023

    const int ln  = lane & 15;
    const int hi  = lane >> 4;
    const int kbo = hi * 8;

    const __bf16* arow = xb + (size_t)(tb * 16 + ln) * D_;
    // 4 weight rows, one per 16-feature tile
    const __bf16* brow = Wt + ((size_t)(m * H_ + h) * DH_ + ln) * D_;

    v8f acc0 = {}, acc1 = {}, acc2 = {}, acc3 = {};
    #pragma unroll 2
    for (int k0 = 0; k0 < D_; k0 += 32) {
        Frag a, b;
        a.u[0] = *(const uint4*)(arow + k0 + kbo);
        a.u[1] = *(const uint4*)(arow + k0 + kbo + 16);

        b.u[0] = *(const uint4*)(brow + k0 + kbo);
        b.u[1] = *(const uint4*)(brow + k0 + kbo + 16);
        acc0 = wmma_bf16(a, b, acc0);
        b.u[0] = *(const uint4*)(brow + 16 * D_ + k0 + kbo);
        b.u[1] = *(const uint4*)(brow + 16 * D_ + k0 + kbo + 16);
        acc1 = wmma_bf16(a, b, acc1);
        b.u[0] = *(const uint4*)(brow + 32 * D_ + k0 + kbo);
        b.u[1] = *(const uint4*)(brow + 32 * D_ + k0 + kbo + 16);
        acc2 = wmma_bf16(a, b, acc2);
        b.u[0] = *(const uint4*)(brow + 48 * D_ + k0 + kbo);
        b.u[1] = *(const uint4*)(brow + 48 * D_ + k0 + kbo + 16);
        acc3 = wmma_bf16(a, b, acc3);
    }

    const float* bias = (m == 0) ? bq : (m == 1) ? bk : bv;
    const float bv0 = bias[h * DH_ +  0 + ln];
    const float bv1 = bias[h * DH_ + 16 + ln];
    const float bv2 = bias[h * DH_ + 32 + ln];
    const float bv3 = bias[h * DH_ + 48 + ln];
    #pragma unroll
    for (int r = 0; r < 8; ++r) {
        acc0[r] += bv0; acc1[r] += bv1; acc2[r] += bv2; acc3[r] += bv3;
    }

    const int    bidx = tb >> 6;
    const int    n0   = (tb & 63) * 16;
    const size_t bh   = (size_t)bidx * H_ + h;

    if (m < 2) {
        // row-major [bh][n][64]
        __bf16* dst = ((m == 0) ? qbuf : kbuf) + (bh * N_ + n0) * DH_ + ln;
        #pragma unroll
        for (int r = 0; r < 8; ++r) {
            const size_t ro = (size_t)(r + 8 * hi) * DH_;
            dst[ro +  0] = (__bf16)acc0[r];
            dst[ro + 16] = (__bf16)acc1[r];
            dst[ro + 32] = (__bf16)acc2[r];
            dst[ro + 48] = (__bf16)acc3[r];
        }
    } else {
        // transposed [bh][64][n]; per lane 8 tokens contiguous -> b128 stores
        Pack8 p0, p1, p2, p3;
        #pragma unroll
        for (int r = 0; r < 8; ++r) {
            p0.h[r] = (__bf16)acc0[r]; p1.h[r] = (__bf16)acc1[r];
            p2.h[r] = (__bf16)acc2[r]; p3.h[r] = (__bf16)acc3[r];
        }
        __bf16* dst = vbuf + (bh * DH_ + ln) * N_ + n0 + 8 * hi;
        *(uint4*)(dst +  0 * N_) = p0.u;
        *(uint4*)(dst + 16 * N_) = p1.u;
        *(uint4*)(dst + 32 * N_) = p2.u;
        *(uint4*)(dst + 48 * N_) = p3.u;
    }
}

// ---------------------------------------------------------------------------
// Flash attention: 8 waves per block share one (b,h); K/V chunks are double-
// buffered in LDS via async global->LDS copies. Per 32-key chunk and wave:
// 4 score WMMAs, online softmax, P restaged via per-wave LDS, 4 PV WMMAs +
// 1 ones-WMMA for the row-sum.
// ---------------------------------------------------------------------------
__global__ void __launch_bounds__(256) flash_attn_kernel(
    const __bf16* __restrict__ qbuf, const __bf16* __restrict__ kbuf,
    const __bf16* __restrict__ vbuf, float* __restrict__ out)
{
    __shared__ __bf16 ktile[2][32 * 64];   // [key][feature]
    __shared__ __bf16 vtile[2][64 * 32];   // [feature][key]
    __shared__ __bf16 ptile[8][16 * 32];   // per-wave P staging

    const int tid  = threadIdx.x;
    const int lane = tid & 31;
    const int wv   = tid >> 5;
    const int bh   = blockIdx.x >> 3;              // 0..191
    const int q0   = ((blockIdx.x & 7) * 8 + wv) * 16;

    const int ln  = lane & 15;
    const int hi  = lane >> 4;
    const int kbo = hi * 8;

    // cooperative-copy source pointers / LDS offsets for this thread
    const int krow = tid >> 3, ksub = (tid & 7) * 8;     // K: 32 rows x 64 feats
    const int vft  = tid >> 2, vsub = (tid & 3) * 8;     // V: 64 feats x 32 keys
    const __bf16* ksrc = kbuf + ((size_t)bh * N_ + krow) * DH_ + ksub;
    const __bf16* vsrc = vbuf + ((size_t)bh * DH_ + vft) * N_ + vsub;

    // prologue: async-load chunk 0 into buffer 0
    async_copy16(&ktile[0][krow * 64 + ksub], ksrc);
    async_copy16(&vtile[0][vft * 32 + vsub], vsrc);

    // Q fragments (K = 0..31 and 32..63)
    const __bf16* qrow = qbuf + ((size_t)bh * N_ + q0 + ln) * DH_;
    Frag qa0, qa1;
    qa0.u[0] = *(const uint4*)(qrow + kbo);
    qa0.u[1] = *(const uint4*)(qrow + kbo + 16);
    qa1.u[0] = *(const uint4*)(qrow + 32 + kbo);
    qa1.u[1] = *(const uint4*)(qrow + 32 + kbo + 16);

    // all-ones B fragment for free row-sum accumulation
    Frag ones;
    #pragma unroll
    for (int i = 0; i < 16; ++i) ones.v[i] = (__bf16)1.0f;

    float mrow[8];
    v8f O0 = {}, O1 = {}, O2 = {}, O3 = {}, Ol = {};
    #pragma unroll
    for (int r = 0; r < 8; ++r) mrow[r] = -1e30f;

    __bf16* sm = &ptile[wv][0];
    int cur = 0;

    for (int j = 0; j < N_; j += 32) {
        wait_async0();         // own async loads of chunk j complete
        __syncthreads();       // all waves' loads complete -> LDS chunk ready
        if (j + 32 < N_) {     // overlap next chunk copy with compute
            async_copy16(&ktile[cur ^ 1][krow * 64 + ksub], ksrc + (size_t)(j + 32) * DH_);
            async_copy16(&vtile[cur ^ 1][vft * 32 + vsub], vsrc + (j + 32));
        }

        const __bf16* kt = &ktile[cur][0];
        const __bf16* vt = &vtile[cur][0];

        // ---- scores for keys [j, j+32): two 16x16 f32 tiles ----
        v8f s0 = {}, s1 = {};
        {
            Frag b;
            b.u[0] = *(const uint4*)(kt + ln * 64 + kbo);
            b.u[1] = *(const uint4*)(kt + ln * 64 + kbo + 16);
            s0 = wmma_bf16(qa0, b, s0);
            b.u[0] = *(const uint4*)(kt + ln * 64 + 32 + kbo);
            b.u[1] = *(const uint4*)(kt + ln * 64 + 32 + kbo + 16);
            s0 = wmma_bf16(qa1, b, s0);
            b.u[0] = *(const uint4*)(kt + (16 + ln) * 64 + kbo);
            b.u[1] = *(const uint4*)(kt + (16 + ln) * 64 + kbo + 16);
            s1 = wmma_bf16(qa0, b, s1);
            b.u[0] = *(const uint4*)(kt + (16 + ln) * 64 + 32 + kbo);
            b.u[1] = *(const uint4*)(kt + (16 + ln) * 64 + 32 + kbo + 16);
            s1 = wmma_bf16(qa1, b, s1);
        }

        // ---- online softmax (row M = r + 8*hi lives in a 16-lane half) ----
        #pragma unroll
        for (int r = 0; r < 8; ++r) {
            float a = s0[r] * 0.125f;
            float b = s1[r] * 0.125f;
            float rm = fmaxf(a, b);
            rm = fmaxf(rm, __shfl_xor(rm, 1));
            rm = fmaxf(rm, __shfl_xor(rm, 2));
            rm = fmaxf(rm, __shfl_xor(rm, 4));
            rm = fmaxf(rm, __shfl_xor(rm, 8));
            float mn = fmaxf(mrow[r], rm);
            float sc = __expf(mrow[r] - mn);
            mrow[r] = mn;
            float p0 = __expf(a - mn);
            float p1 = __expf(b - mn);
            sm[(r + 8 * hi) * 32 + ln]      = (__bf16)p0;
            sm[(r + 8 * hi) * 32 + 16 + ln] = (__bf16)p1;
            O0[r] *= sc; O1[r] *= sc; O2[r] *= sc; O3[r] *= sc; Ol[r] *= sc;
        }

        // ---- re-fragment P as A-matrix from per-wave LDS ----
        Frag pa;
        pa.u[0] = *(const uint4*)(sm + ln * 32 + kbo);
        pa.u[1] = *(const uint4*)(sm + ln * 32 + kbo + 16);

        // ---- P @ V: 4 feature tiles + ones tile for rowsum ----
        Frag vb;
        vb.u[0] = *(const uint4*)(vt + ln * 32 + kbo);
        vb.u[1] = *(const uint4*)(vt + ln * 32 + kbo + 16);
        O0 = wmma_bf16(pa, vb, O0);
        vb.u[0] = *(const uint4*)(vt + (16 + ln) * 32 + kbo);
        vb.u[1] = *(const uint4*)(vt + (16 + ln) * 32 + kbo + 16);
        O1 = wmma_bf16(pa, vb, O1);
        vb.u[0] = *(const uint4*)(vt + (32 + ln) * 32 + kbo);
        vb.u[1] = *(const uint4*)(vt + (32 + ln) * 32 + kbo + 16);
        O2 = wmma_bf16(pa, vb, O2);
        vb.u[0] = *(const uint4*)(vt + (48 + ln) * 32 + kbo);
        vb.u[1] = *(const uint4*)(vt + (48 + ln) * 32 + kbo + 16);
        O3 = wmma_bf16(pa, vb, O3);
        Ol = wmma_bf16(pa, ones, Ol);

        cur ^= 1;
    }

    // ---- finalize: divide by rowsum, store fp32 into [B, N, H*DH] ----
    const int bb = bh / H_;
    const int hh = bh % H_;
    float* obase = out + ((size_t)bb * N_ + q0) * (H_ * DH_) + hh * DH_ + ln;
    #pragma unroll
    for (int r = 0; r < 8; ++r) {
        const float  inv  = 1.0f / Ol[r];
        const size_t roff = (size_t)(r + 8 * hi) * (H_ * DH_);
        obase[roff +  0] = O0[r] * inv;
        obase[roff + 16] = O1[r] * inv;
        obase[roff + 32] = O2[r] * inv;
        obase[roff + 48] = O3[r] * inv;
    }
}

// ---------------------------------------------------------------------------
extern "C" void kernel_launch(void* const* d_in, const int* in_sizes, int n_in,
                              void* d_out, int out_size, void* d_ws, size_t ws_size,
                              hipStream_t stream) {
    const float* x  = (const float*)d_in[0];
    const float* Wq = (const float*)d_in[1];
    const float* Wk = (const float*)d_in[2];
    const float* Wv = (const float*)d_in[3];
    const float* bq = (const float*)d_in[4];
    const float* bk = (const float*)d_in[5];
    const float* bv = (const float*)d_in[6];
    float* out = (float*)d_out;

    char* ws = (char*)d_ws;
    const size_t xb_bytes = (size_t)B_ * N_ * D_ * 2;            // 25,165,824
    const size_t wt_bytes = (size_t)3 * H_ * DH_ * D_ * 2;       //  3,538,944
    const size_t qk_elems = (size_t)BH_ * N_ * DH_;              // 12,582,912

    __bf16* xb   = (__bf16*)ws;
    __bf16* Wt   = (__bf16*)(ws + xb_bytes);
    __bf16* qbuf = (__bf16*)(ws + xb_bytes + wt_bytes);
    __bf16* kbuf = qbuf + qk_elems;
    __bf16* vbuf = kbuf + qk_elems;

    const int n4 = B_ * N_ * D_ / 4;
    prep_x_kernel<<<(n4 + 255) / 256, 256, 0, stream>>>(x, xb, n4);

    const int wtot = 3 * H_ * DH_ * D_;
    prep_w_kernel<<<(wtot + 255) / 256, 256, 0, stream>>>(Wq, Wk, Wv, Wt, wtot);

    // 1024 token blocks * 3 matrices * 12 heads = 36864 jobs, 8 waves/block
    proj_qkv_kernel<<<36864 / 8, 256, 0, stream>>>(xb, Wt, bq, bk, bv,
                                                   qbuf, kbuf, vbuf);

    // 192 (b,h) * 8 query-groups = 1536 blocks; 8 waves of 16 queries each
    flash_attn_kernel<<<192 * 8, 256, 0, stream>>>(qbuf, kbuf, vbuf, out);
}
